// mvPAM_Module_mask_cascade_26113401159782
// MI455X (gfx1250) — compile-verified, matrix operand order checked by reading, not run on hardware
//
#include <hip/hip_runtime.h>
#include <hip/hip_bf16.h>
#include <math.h>

#define DEVI __device__ __forceinline__

typedef __attribute__((ext_vector_type(16))) __bf16 bf16x16;
typedef __attribute__((ext_vector_type(8)))  __bf16 bf16x8;
typedef __attribute__((ext_vector_type(8)))  float  f32x8;
typedef __attribute__((ext_vector_type(4))) unsigned int u32x4;
typedef __attribute__((ext_vector_type(8))) int i32x8;
typedef __attribute__((ext_vector_type(4))) int i32x4;

constexpr int Bn = 4, Cn = 512, Nn = 4096, Dn = 64;
constexpr int WPB = 8;          // waves per block
constexpr int TPB = WPB * 32;   // 256 threads

#if __has_builtin(__builtin_amdgcn_tensor_load_to_lds)
#define HAVE_TDM 1
#if __has_include(<hip/amd_detail/amd_gfx1250_TDM.h>)
#define TDM_6ARG 1
#else
#define TDM_6ARG 0
#endif
#else
#define HAVE_TDM 0
#endif

DEVI f32x8 zero8() {
  f32x8 z;
#pragma unroll
  for (int i = 0; i < 8; ++i) z[i] = 0.f;
  return z;
}

DEVI f32x8 wmma_bf16(bf16x16 a, bf16x16 b, f32x8 c) {
  // D = A(16x32 bf16) * B(32x16 bf16) + C(16x16 f32)
  return __builtin_amdgcn_wmma_f32_16x16x32_bf16(false, a, false, b, (short)0, c,
                                                 false, false);
}

// K-contiguous fragment: element (r, k) at base[r*stride + k].
// Lane layout (wave32): r = lane&15, h = lane>>4;
//   elems 0..7  = K[h*8 .. h*8+7], elems 8..15 = K[16+h*8 .. 16+h*8+7]
DEVI bf16x16 frag_ld(const __bf16* __restrict__ base, long stride, int r, int h) {
  const __bf16* p = base + (long)r * stride;
  bf16x8 lo = *(const bf16x8*)(p + h * 8);
  bf16x8 hi = *(const bf16x8*)(p + 16 + h * 8);
  bf16x16 f;
#pragma unroll
  for (int j = 0; j < 8; ++j) { f[j] = lo[j]; f[8 + j] = hi[j]; }
  return f;
}

// Same but source is f32 row-major (weights) -> convert to bf16
DEVI bf16x16 frag_ld_f32(const float* __restrict__ base, long stride, int r, int h) {
  const float* p = base + (long)r * stride;
  bf16x16 f;
#pragma unroll
  for (int j = 0; j < 8; ++j) {
    f[j]     = (__bf16)p[h * 8 + j];
    f[8 + j] = (__bf16)p[16 + h * 8 + j];
  }
  return f;
}

#if HAVE_TDM
// Issue a TDM 2D tile load (bf16 elements): tile_d1 rows x tile_d0 cols,
// row stride = stride_elems, into LDS at byte offset lds_off.
DEVI void tdm_load_2d_bf16(unsigned lds_off, unsigned long long gaddr,
                           unsigned tile_d0, unsigned tile_d1,
                           unsigned long long stride_elems,
                           unsigned tensor_d0, unsigned tensor_d1) {
  u32x4 g0;
  g0[0] = 1u;                                     // count=1, user descriptor
  g0[1] = lds_off;                                // lds_addr (bytes)
  g0[2] = (unsigned)(gaddr & 0xffffffffull);      // global_addr[31:0]
  g0[3] = (unsigned)((gaddr >> 32) & 0x01ffffffull) | (2u << 30);  // addr[56:32], type=2
  i32x8 g1;
  g1[0] = (int)(1u << 16);                        // wg_mask=0, data_size=1 (2 bytes)
  g1[1] = (int)((tensor_d0 & 0xffffu) << 16);     // tensor_dim0[15:0]
  g1[2] = (int)((tensor_d0 >> 16) | ((tensor_d1 & 0xffffu) << 16));
  g1[3] = (int)((tensor_d1 >> 16) | (tile_d0 << 16));   // tile_dim0
  g1[4] = (int)(tile_d1 & 0xffffu);               // tile_dim1 (tile_dim2=0)
  g1[5] = (int)(unsigned)(stride_elems & 0xffffffffull);        // dim0 stride lo
  g1[6] = (int)(unsigned)((stride_elems >> 32) & 0xffffull);    // stride hi, dim1_stride=0
  g1[7] = 0;
  i32x4 z4;
#pragma unroll
  for (int i = 0; i < 4; ++i) z4[i] = 0;
#if TDM_6ARG
  i32x8 z8;
#pragma unroll
  for (int i = 0; i < 8; ++i) z8[i] = 0;
  __builtin_amdgcn_tensor_load_to_lds(g0, g1, z4, z4, z8, 0);
#else
  __builtin_amdgcn_tensor_load_to_lds(g0, g1, z4, z4, 0);
#endif
}
#endif

// ---------------------------------------------------------------- cvt f32->bf16
__global__ void cvt_kernel(const float* __restrict__ x, __bf16* __restrict__ y,
                           long n) {
  long i = ((long)blockIdx.x * blockDim.x + threadIdx.x) * 4;
  if (i + 3 < n) {
    float4 v = *(const float4*)(x + i);
    y[i + 0] = (__bf16)v.x; y[i + 1] = (__bf16)v.y;
    y[i + 2] = (__bf16)v.z; y[i + 3] = (__bf16)v.w;
  }
}

// ------------------------------------------------- Q/K projections (WMMA, d=64)
__global__ void proj_qk_kernel(const float* __restrict__ x,
                               const float* __restrict__ wq, const float* __restrict__ bq,
                               const float* __restrict__ wk, const float* __restrict__ bk,
                               __bf16* __restrict__ Qb, __bf16* __restrict__ Kb) {
  int lane = threadIdx.x & 31;
  int w = blockIdx.x * WPB + (threadIdx.x >> 5);
  int b = w / (Nn / 16);
  int m0 = (w % (Nn / 16)) * 16;
  int r = lane & 15, h = lane >> 4;
  const float* xb = x + (long)b * Cn * Nn;

  f32x8 aq[4], ak[4];
#pragma unroll
  for (int t = 0; t < 4; ++t) { aq[t] = zero8(); ak[t] = zero8(); }

  for (int k0 = 0; k0 < Cn; k0 += 32) {
    const float* xp = xb + (long)k0 * Nn + (m0 + r);
    bf16x16 a;
#pragma unroll
    for (int j = 0; j < 8; ++j) {
      a[j]     = (__bf16)xp[(long)(h * 8 + j) * Nn];
      a[8 + j] = (__bf16)xp[(long)(16 + h * 8 + j) * Nn];
    }
#pragma unroll
    for (int t = 0; t < 4; ++t) {
      bf16x16 fq = frag_ld_f32(wq + (long)(t * 16) * Cn + k0, Cn, r, h);
      aq[t] = wmma_bf16(a, fq, aq[t]);
      bf16x16 fk = frag_ld_f32(wk + (long)(t * 16) * Cn + k0, Cn, r, h);
      ak[t] = wmma_bf16(a, fk, ak[t]);
    }
  }
#pragma unroll
  for (int t = 0; t < 4; ++t) {
    int dd = t * 16 + r;
    float biq = bq[dd], bik = bk[dd];
#pragma unroll
    for (int i = 0; i < 8; ++i) {
      long row = (long)b * Nn + m0 + i + 8 * h;
      Qb[row * Dn + dd] = (__bf16)(aq[t][i] + biq);
      Kb[row * Dn + dd] = (__bf16)(ak[t][i] + bik);
    }
  }
}

// ------------------------------------------------------- energy = Q @ K^T (bf16)
__global__ void energy_kernel(const __bf16* __restrict__ Qb,
                              const __bf16* __restrict__ Kb,
                              __bf16* __restrict__ E) {
  int lane = threadIdx.x & 31;
  int w = blockIdx.x * WPB + (threadIdx.x >> 5);
  int per_b = (Nn / 16) * (Nn / 64);
  int b = w / per_b;
  int rem = w % per_b;
  int m0 = (rem / (Nn / 64)) * 16;
  int n0 = (rem % (Nn / 64)) * 64;
  int r = lane & 15, h = lane >> 4;

  f32x8 acc[4];
#pragma unroll
  for (int t = 0; t < 4; ++t) acc[t] = zero8();

  const __bf16* Qp = Qb + ((long)b * Nn + m0) * Dn;
#pragma unroll
  for (int k0 = 0; k0 < Dn; k0 += 32) {
    bf16x16 a = frag_ld(Qp + k0, Dn, r, h);
#pragma unroll
    for (int t = 0; t < 4; ++t) {
      bf16x16 bb = frag_ld(Kb + ((long)b * Nn + n0 + t * 16) * Dn + k0, Dn, r, h);
      acc[t] = wmma_bf16(a, bb, acc[t]);
    }
  }
#pragma unroll
  for (int t = 0; t < 4; ++t)
#pragma unroll
    for (int i = 0; i < 8; ++i)
      E[((long)b * Nn + m0 + i + 8 * h) * Nn + n0 + t * 16 + r] = (__bf16)acc[t][i];
}

// ------------------------------ per-row softmax stats for 3 masks (one E pass)
__global__ void stats_kernel(const __bf16* __restrict__ E,
                             float* __restrict__ mxS, float* __restrict__ ivS) {
  int lane = threadIdx.x & 31;
  int w = blockIdx.x * WPB + (threadIdx.x >> 5);  // row id b*Nn+m
  int m = w % Nn;
  const __bf16* Er = E + (long)w * Nn;
  float mx[3] = {-3.0e38f, -3.0e38f, -3.0e38f};
  float sm[3] = {0.f, 0.f, 0.f};
  for (int n = lane; n < Nn; n += 32) {
    float e = (float)Er[n];
    int dd = n - m; if (dd < 0) dd = -dd;
#pragma unroll
    for (int s = 0; s < 3; ++s) {
      int win = (s == 0) ? 256 : (s == 1 ? 1024 : Nn);
      if (dd <= win) {
        if (e > mx[s]) { sm[s] = sm[s] * __expf(mx[s] - e) + 1.f; mx[s] = e; }
        else            sm[s] += __expf(e - mx[s]);
      }
    }
  }
#pragma unroll
  for (int s = 0; s < 3; ++s) {
    float mxv = mx[s], smv = sm[s];
    for (int o = 16; o >= 1; o >>= 1) {
      float m2 = __shfl_xor(mxv, o, 32);
      float s2 = __shfl_xor(smv, o, 32);
      float M = fmaxf(mxv, m2);
      smv = smv * __expf(mxv - M) + s2 * __expf(m2 - M);
      mxv = M;
    }
    if (lane == 0) {
      mxS[s * (Bn * Nn) + w] = mxv;
      ivS[s * (Bn * Nn) + w] = 1.f / smv;
    }
  }
}

// --------------------------------------- 1x1 conv: O = W(CxC,f32) @ X(CxN,bf16)
// Block: one (b, n0..n0+64) strip, 8 waves cover 128 output channels.
// X tile (32ci x 64n bf16, 4KB) staged to LDS by TDM, double-buffered.
__global__ void conv_kernel(const __bf16* __restrict__ Xin,
                            const float* __restrict__ W,
                            const float* __restrict__ bias,
                            __bf16* __restrict__ Out) {
  int lane = threadIdx.x & 31;
  int widx = threadIdx.x >> 5;
  int per_b = (Cn / 128) * (Nn / 64);            // 4 * 64 = 256 blocks per batch
  int b  = blockIdx.x / per_b;
  int rem = blockIdx.x % per_b;
  int co0 = (rem / (Nn / 64)) * 128 + widx * 16;
  int n0  = (rem % (Nn / 64)) * 64;
  int r = lane & 15, h = lane >> 4;

  f32x8 acc[4];
#pragma unroll
  for (int t = 0; t < 4; ++t) acc[t] = zero8();

  const __bf16* Xb = Xin + (long)b * Cn * Nn + n0;

#if HAVE_TDM
  __shared__ __bf16 ldsX[2][32][64];
  unsigned lds_base = (unsigned)(unsigned long long)(&ldsX[0][0][0]);
  const int nk = Cn / 32;
  if (widx == 0) {
    tdm_load_2d_bf16(lds_base, (unsigned long long)(unsigned long)(const void*)Xb,
                     64u, 32u, (unsigned long long)Nn, (unsigned)Nn, (unsigned)Cn);
  }
  for (int i = 0; i < nk; ++i) {
    int k0 = i * 32;
    if (widx == 0) __builtin_amdgcn_s_wait_tensorcnt(0);
    __syncthreads();                               // tile i ready for all waves
    if (widx == 0 && i + 1 < nk) {
      tdm_load_2d_bf16(lds_base + ((unsigned)((i + 1) & 1)) * 4096u,
                       (unsigned long long)(unsigned long)(const void*)(Xb + (long)(k0 + 32) * Nn),
                       64u, 32u, (unsigned long long)Nn, (unsigned)Nn, (unsigned)Cn);
    }
    bf16x16 a = frag_ld_f32(W + (long)co0 * Cn + k0, Cn, r, h);
    const __bf16 (*tile)[64] = ldsX[i & 1];
#pragma unroll
    for (int t = 0; t < 4; ++t) {
      bf16x16 bb;
#pragma unroll
      for (int j = 0; j < 8; ++j) {
        bb[j]     = tile[h * 8 + j][t * 16 + r];
        bb[8 + j] = tile[16 + h * 8 + j][t * 16 + r];
      }
      acc[t] = wmma_bf16(a, bb, acc[t]);
    }
    __syncthreads();                               // done reading before overwrite
  }
#else
  for (int k0 = 0; k0 < Cn; k0 += 32) {
    bf16x16 a = frag_ld_f32(W + (long)co0 * Cn + k0, Cn, r, h);
#pragma unroll
    for (int t = 0; t < 4; ++t) {
      const __bf16* xp = Xb + (long)k0 * Nn + t * 16 + r;
      bf16x16 bb;
#pragma unroll
      for (int j = 0; j < 8; ++j) {
        bb[j]     = xp[(long)(h * 8 + j) * Nn];
        bb[8 + j] = xp[(long)(16 + h * 8 + j) * Nn];
      }
      acc[t] = wmma_bf16(a, bb, acc[t]);
    }
  }
#endif

#pragma unroll
  for (int t = 0; t < 4; ++t)
#pragma unroll
    for (int i = 0; i < 8; ++i) {
      int co = co0 + i + 8 * h;
      Out[((long)b * Cn + co) * Nn + n0 + t * 16 + r] = (__bf16)(acc[t][i] + bias[co]);
    }
}

// --------- attention apply: out[c,m] = inv[m] * sum_n exp(E[m,n]-mx[m]) V[c,n]
template <int WIN, bool FINAL>
__global__ void apply_kernel(const __bf16* __restrict__ E,
                             const float* __restrict__ mxS,
                             const float* __restrict__ ivS,
                             const __bf16* __restrict__ V,
                             __bf16* __restrict__ Outb,
                             float* __restrict__ Outf,
                             const float* __restrict__ gamma,
                             const float* __restrict__ xres) {
  int lane = threadIdx.x & 31;
  int w = blockIdx.x * WPB + (threadIdx.x >> 5);
  int per_b = (Nn / 16) * (Cn / 64);
  int b = w / per_b;
  int rem = w % per_b;
  int m0 = (rem / (Cn / 64)) * 16;
  int c0 = (rem % (Cn / 64)) * 64;
  int r = lane & 15, h = lane >> 4;
  int mrow = m0 + r;
  long rowidx = (long)b * Nn + mrow;
  float mxv = mxS[rowidx];
  float ivv = ivS[rowidx];
  const __bf16* Er = E + rowidx * Nn;

  int nlo = 0, nhi = Nn;
  if (WIN < Nn) {
    nlo = m0 - WIN; if (nlo < 0) nlo = 0; nlo &= ~31;
    nhi = m0 + 16 + WIN; if (nhi > Nn) nhi = Nn; nhi = (nhi + 31) & ~31;
  }

  f32x8 acc[4];
#pragma unroll
  for (int t = 0; t < 4; ++t) acc[t] = zero8();

  for (int n0 = nlo; n0 < nhi; n0 += 32) {
    if (n0 + 128 < nhi) __builtin_prefetch(Er + n0 + 128, 0, 1);
    bf16x8 e0 = *(const bf16x8*)(Er + n0 + h * 8);
    bf16x8 e1 = *(const bf16x8*)(Er + n0 + 16 + h * 8);
    bf16x16 a;
#pragma unroll
    for (int j = 0; j < 8; ++j) {
      int nA = n0 + h * 8 + j;
      int nB = n0 + 16 + h * 8 + j;
      float p0 = __expf((float)e0[j] - mxv) * ivv;
      float p1 = __expf((float)e1[j] - mxv) * ivv;
      if (WIN < Nn) {
        int d0 = nA - mrow; if (d0 < 0) d0 = -d0;
        int d1 = nB - mrow; if (d1 < 0) d1 = -d1;
        if (d0 > WIN) p0 = 0.f;
        if (d1 > WIN) p1 = 0.f;
      }
      a[j]     = (__bf16)p0;
      a[8 + j] = (__bf16)p1;
    }
#pragma unroll
    for (int t = 0; t < 4; ++t) {
      bf16x16 bb = frag_ld(V + ((long)b * Cn + c0 + t * 16) * Nn + n0, Nn, r, h);
      acc[t] = wmma_bf16(a, bb, acc[t]);
    }
  }

#pragma unroll
  for (int t = 0; t < 4; ++t) {
    long c = c0 + t * 16 + r;
    long obase = ((long)b * Cn + c) * Nn + m0 + 8 * h;
    if (FINAL) {
      float g = gamma[0];
#pragma unroll
      for (int i = 0; i < 8; ++i)
        Outf[obase + i] = g * acc[t][i] + xres[obase + i];
    } else {
      bf16x8 ov;
#pragma unroll
      for (int i = 0; i < 8; ++i) ov[i] = (__bf16)acc[t][i];
      *(bf16x8*)(Outb + obase) = ov;
    }
  }
}

// ---------------------------------------------------------------------- launch
extern "C" void kernel_launch(void* const* d_in, const int* in_sizes, int n_in,
                              void* d_out, int out_size, void* d_ws, size_t ws_size,
                              hipStream_t stream) {
  (void)in_sizes; (void)n_in; (void)out_size; (void)ws_size;
  const float* x    = (const float*)d_in[0];
  const float* wq   = (const float*)d_in[1];
  const float* bq   = (const float*)d_in[2];
  const float* wk   = (const float*)d_in[3];
  const float* bk   = (const float*)d_in[4];
  const float* wv0  = (const float*)d_in[5];
  const float* bv0  = (const float*)d_in[6];
  const float* wv1  = (const float*)d_in[7];
  const float* bv1  = (const float*)d_in[8];
  const float* wv2  = (const float*)d_in[9];
  const float* bv2  = (const float*)d_in[10];
  const float* gm   = (const float*)d_in[11];
  float* out = (float*)d_out;

  char* wsp = (char*)d_ws;
  auto alloc = [&](size_t bytes) {
    char* p = wsp; wsp += (bytes + 255) & ~(size_t)255; return p;
  };
  const size_t cxn = (size_t)Bn * Cn * Nn * sizeof(__bf16);
  __bf16* Xb  = (__bf16*)alloc(cxn);
  __bf16* V0  = (__bf16*)alloc(cxn);
  __bf16* O0  = (__bf16*)alloc(cxn);
  __bf16* O0c = (__bf16*)alloc(cxn);
  __bf16* O1  = (__bf16*)alloc(cxn);
  __bf16* O1c = (__bf16*)alloc(cxn);
  __bf16* Qb  = (__bf16*)alloc((size_t)Bn * Nn * Dn * sizeof(__bf16));
  __bf16* Kb  = (__bf16*)alloc((size_t)Bn * Nn * Dn * sizeof(__bf16));
  __bf16* E   = (__bf16*)alloc((size_t)Bn * Nn * Nn * sizeof(__bf16));
  float*  mxS = (float*)alloc((size_t)3 * Bn * Nn * sizeof(float));
  float*  ivS = (float*)alloc((size_t)3 * Bn * Nn * sizeof(float));

  const long total = (long)Bn * Cn * Nn;
  const int g_cvt    = (int)(total / (TPB * 4));
  const int g_proj   = Bn * (Nn / 16) / WPB;
  const int g_energy = Bn * (Nn / 16) * (Nn / 64) / WPB;
  const int g_stats  = Bn * Nn / WPB;
  const int g_conv   = Bn * (Cn / 128) * (Nn / 64);   // 1 block = 128co x 64n
  const int g_apply  = Bn * (Nn / 16) * (Cn / 64) / WPB;

  cvt_kernel<<<g_cvt, TPB, 0, stream>>>(x, Xb, total);
  proj_qk_kernel<<<g_proj, TPB, 0, stream>>>(x, wq, bq, wk, bk, Qb, Kb);
  energy_kernel<<<g_energy, TPB, 0, stream>>>(Qb, Kb, E);
  stats_kernel<<<g_stats, TPB, 0, stream>>>(E, mxS, ivS);

  conv_kernel<<<g_conv, TPB, 0, stream>>>(Xb, wv0, bv0, V0);
  apply_kernel<256, false><<<g_apply, TPB, 0, stream>>>(
      E, mxS, ivS, V0, O0, nullptr, nullptr, nullptr);
  conv_kernel<<<g_conv, TPB, 0, stream>>>(O0, wv1, bv1, O0c);
  apply_kernel<1024, false><<<g_apply, TPB, 0, stream>>>(
      E, mxS + (size_t)Bn * Nn, ivS + (size_t)Bn * Nn, O0c, O1, nullptr, nullptr, nullptr);
  conv_kernel<<<g_conv, TPB, 0, stream>>>(O1, wv2, bv2, O1c);
  apply_kernel<Nn, true><<<g_apply, TPB, 0, stream>>>(
      E, mxS + (size_t)2 * Bn * Nn, ivS + (size_t)2 * Bn * Nn, O1c, nullptr, out, gm, x);
}
